// FormerBlock_21088289423634
// MI455X (gfx1250) — matrix operand order repaired and checked by reading, not compiled, not run on hardware
//
#include <hip/hip_runtime.h>
#include <hip/hip_bf16.h>
#include <math.h>

#define B_    2
#define L_    384
#define D_    576
#define H_    8
#define DK3_  24
#define C3_   192
#define PAIR_ 128
#define EXP_  1152
#define KCF_  1728            /* im2col K for conv-FFN expand: 576*3 */
#define EPS_  1e-6f
#define DIVQ  8.48528137423857f   /* sqrt(72) */

typedef __attribute__((ext_vector_type(16))) _Float16 v16h;
typedef __attribute__((ext_vector_type(8)))  float    v8f;
typedef __attribute__((ext_vector_type(4)))  unsigned int u32x4;
typedef __attribute__((ext_vector_type(4)))  int      i32x4;
typedef __attribute__((ext_vector_type(8)))  int      i32x8;

#if __has_builtin(__builtin_amdgcn_tensor_load_to_lds) && \
    __has_builtin(__builtin_amdgcn_s_wait_tensorcnt)
#define HAVE_TDM 1
#else
#define HAVE_TDM 0
#endif

// ---------------------------------------------------------------- helpers
__device__ __forceinline__ float geluf(float x) {
  return 0.5f * x * (1.f + erff(x * 0.70710678118654752f));
}
__device__ __forceinline__ float sigm(float x) { return 1.f / (1.f + expf(-x)); }

__device__ __forceinline__ float block_sum(float v, float* red) {
  int t = threadIdx.x; red[t] = v; __syncthreads();
  for (int off = blockDim.x >> 1; off > 0; off >>= 1) {
    if (t < off) red[t] += red[t + off];
    __syncthreads();
  }
  float r = red[0]; __syncthreads(); return r;
}
__device__ __forceinline__ float block_max(float v, float* red) {
  int t = threadIdx.x; red[t] = v; __syncthreads();
  for (int off = blockDim.x >> 1; off > 0; off >>= 1) {
    if (t < off) red[t] = fmaxf(red[t], red[t + off]);
    __syncthreads();
  }
  float r = red[0]; __syncthreads(); return r;
}

// ---- WMMA fragment packing (CDNA5 ISA 7.12.2, wave32) ----
__device__ __forceinline__ v16h load_a_frag(const _Float16* rowbase, int rowStride, int lane) {
  int m = lane & 15;
  int kadd = (lane >> 4) << 4;
  v16h a;
#pragma unroll
  for (int h2 = 0; h2 < 16; ++h2) {
    int k = h2 + kadd;
    a[h2] = rowbase[m * rowStride + k];
  }
  return a;
}
__device__ __forceinline__ v16h load_b_frag(const _Float16* colbase, int colStride, int lane) {
  int n = lane & 15;
  int kadd = (lane >> 4) << 4;
  v16h b;
#pragma unroll
  for (int h2 = 0; h2 < 16; ++h2) b[h2] = colbase[n * colStride + h2 + kadd];
  return b;
}
__device__ __forceinline__ v16h load_b_frag_f32(const float* colbase, int colStride, int lane) {
  int n = lane & 15;
  int kadd = (lane >> 4) << 4;
  v16h b;
#pragma unroll
  for (int h2 = 0; h2 < 16; ++h2) b[h2] = (_Float16)colbase[n * colStride + h2 + kadd];
  return b;
}

// ---- Tensor Data Mover: flat 1-D global(f16) -> LDS stage (ISA 08, D# per §8.3/8.4) ----
#if HAVE_TDM
__device__ __forceinline__ void tdm_stage(_Float16* ldst, const _Float16* gsrc,
                                          unsigned halves) {
  unsigned long long ga = (unsigned long long)(const void*)gsrc;
  unsigned lds_off = (unsigned)(unsigned long long)(void*)ldst;  // LDS aperture: addr[31:0]
  u32x4 g0;
  g0[0] = 1u;                                          // count=1, user mode, no gather
  g0[1] = lds_off;                                     // lds_addr (bytes)
  g0[2] = (unsigned)(ga & 0xffffffffu);                // global_addr[31:0]
  g0[3] = (unsigned)((ga >> 32) & 0x01ffffffu) | (2u << 30);  // addr[56:32] | type=2
  i32x8 g1;
  g1[0] = (int)(1u << 16);                             // data_size=1 (2 bytes)
  g1[1] = (int)((halves & 0xffffu) << 16);             // tensor_dim0[15:0]
  g1[2] = (int)((halves >> 16) & 0xffffu);             // tensor_dim0[31:16] | tensor_dim1=0
  g1[3] = (int)((halves & 0xffffu) << 16);             // tile_dim0
  g1[4] = 0;                                           // tile_dim1=0 (1-D), tile_dim2=0
  g1[5] = (int)halves;                                 // tensor_dim0_stride[31:0]
  g1[6] = 0;
  g1[7] = 0;
  i32x4 z4 = {0, 0, 0, 0};
#if __clang_major__ >= 23
  i32x8 z8 = {0, 0, 0, 0, 0, 0, 0, 0};
  __builtin_amdgcn_tensor_load_to_lds(g0, g1, z4, z4, z8, 0);
#else
  __builtin_amdgcn_tensor_load_to_lds(g0, g1, z4, z4, 0);
#endif
}
#endif

// ---------------------------------------------------------------- generic WMMA GEMM
// Y[m][n] = epi( sum_k Ah[m][k] * W[n][k] + bias[n] )  (+ res[m][n]*gscale)
__global__ void __launch_bounds__(32)
k_gemm(const _Float16* __restrict__ Ah, int lda,
       const float* __restrict__ W, int ldw,
       const float* __restrict__ bias,
       const float* __restrict__ res, int ldr,
       const float* __restrict__ gscale,
       float* __restrict__ Y, _Float16* __restrict__ Yh, int ldo,
       int K, int act) {
  int m0 = blockIdx.x * 16, n0 = blockIdx.y * 16;
  int lane = threadIdx.x;
  v8f acc = {0.f, 0.f, 0.f, 0.f, 0.f, 0.f, 0.f, 0.f};
  for (int k0 = 0; k0 < K; k0 += 32) {
    if (k0 + 32 < K)
      __builtin_prefetch(W + (size_t)n0 * ldw + k0 + 32, 0, 0);   // global_prefetch_b8
    v16h a = load_a_frag(Ah + (size_t)m0 * lda + k0, lda, lane);
    v16h b = load_b_frag_f32(W + (size_t)n0 * ldw + k0, ldw, lane);
    acc = __builtin_amdgcn_wmma_f32_16x16x32_f16(false, a, false, b, (short)0, acc, false, false);
  }
  int n = n0 + (lane & 15);
  int rb = (lane >> 4) * 8;
  float gs = gscale ? gscale[0] : 1.f;
#pragma unroll
  for (int r = 0; r < 8; ++r) {
    int m = m0 + r + rb;
    float v = acc[r] + (bias ? bias[n] : 0.f);
    if (act == 1) v = geluf(v);
    if (res) v += res[(size_t)m * ldr + n] * gs;
    if (Y)  Y[(size_t)m * ldo + n] = v;
    if (Yh) Yh[(size_t)m * ldo + n] = (_Float16)v;
  }
}

// ---------------------------------------------------------------- f32 -> f16 array convert
__global__ void k_f32h(const float* __restrict__ src, _Float16* __restrict__ dst, int n) {
  int i = blockIdx.x * blockDim.x + threadIdx.x;
  if (i < n) dst[i] = (_Float16)src[i];
}

// ---------------------------------------------------------------- LayerNorm over rows (f32 and/or f16 out)
__global__ void k_ln(const float* __restrict__ in, const float* __restrict__ g,
                     const float* __restrict__ bt, float* __restrict__ out,
                     _Float16* __restrict__ outh, int C) {
  int row = blockIdx.x;
  const float* x = in + (size_t)row * C;
  __shared__ float red[256];
  float s = 0.f, s2 = 0.f;
  for (int c = threadIdx.x; c < C; c += blockDim.x) { float v = x[c]; s += v; s2 += v * v; }
  float tot = block_sum(s, red);
  float tot2 = block_sum(s2, red);
  float mean = tot / C;
  float inv = rsqrtf(tot2 / C - mean * mean + EPS_);
  for (int c = threadIdx.x; c < C; c += blockDim.x) {
    float v = (x[c] - mean) * inv * g[c] + bt[c];
    if (out)  out[(size_t)row * C + c] = v;
    if (outh) outh[(size_t)row * C + c] = (_Float16)v;
  }
}

// ---------------------------------------------------------------- im2col staging for conv-FFN expand
__global__ void k_stage_cf(const float* __restrict__ xln, _Float16* __restrict__ Acf) {
  int bl = blockIdx.x; int bb = bl / L_; int l = bl % L_;
  for (int t = threadIdx.x; t < KCF_; t += blockDim.x) {
    int d = t / 3; int tap = t % 3;
    int ls = l + tap - 1;
    float v = (ls >= 0 && ls < L_) ? xln[((size_t)bb * L_ + ls) * D_ + d] : 0.f;
    Acf[(size_t)bl * KCF_ + t] = (_Float16)v;
  }
}

// ---------------------------------------------------------------- depthwise conv tap (k=3)
__global__ void k_dw(const _Float16* __restrict__ h1h, const float* __restrict__ dw,
                     _Float16* __restrict__ h2h) {
  int bl = blockIdx.x; int bb = bl / L_; int l = bl % L_;
  for (int e = threadIdx.x; e < EXP_; e += blockDim.x) {
    float acc = 0.f;
    for (int tap = 0; tap < 3; ++tap) {
      int ls = l + tap - 1;
      if (ls >= 0 && ls < L_)
        acc += (float)h1h[((size_t)bb * L_ + ls) * EXP_ + e] * dw[e * 3 + tap];
    }
    h2h[(size_t)bl * EXP_ + e] = (_Float16)acc;
  }
}

// ---------------------------------------------------------------- column attention small core (3x3 per head) + gate
__global__ void k_col_small(const float* __restrict__ q4, const float* __restrict__ k4,
                            const float* __restrict__ v4, const float* __restrict__ g4,
                            _Float16* __restrict__ och) {
  int bl = blockIdx.x;
  __shared__ float sc[H_][3][3];
  const float* qr = q4 + (size_t)bl * D_;
  const float* kr = k4 + (size_t)bl * D_;
  const float* vr = v4 + (size_t)bl * D_;
  const float* gr = g4 + (size_t)bl * D_;
  if (threadIdx.x < H_ * 9) {
    int t = threadIdx.x;
    int h = t / 9; int a = (t % 9) / 3; int b2 = t % 3;
    float acc = 0.f;
    for (int d = 0; d < DK3_; ++d)
      acc += qr[a * C3_ + h * DK3_ + d] * kr[b2 * C3_ + h * DK3_ + d];
    sc[h][a][b2] = acc / DIVQ;
  }
  __syncthreads();
  if (threadIdx.x < H_ * 3) {
    int h = threadIdx.x / 3; int a = threadIdx.x % 3;
    float m = fmaxf(sc[h][a][0], fmaxf(sc[h][a][1], sc[h][a][2]));
    float e0 = expf(sc[h][a][0] - m), e1 = expf(sc[h][a][1] - m), e2 = expf(sc[h][a][2] - m);
    float ss = e0 + e1 + e2;
    sc[h][a][0] = e0 / ss; sc[h][a][1] = e1 / ss; sc[h][a][2] = e2 / ss;
  }
  __syncthreads();
  for (int c = threadIdx.x; c < D_; c += blockDim.x) {
    int a = c / C3_; int rem = c % C3_; int h = rem / DK3_; int d = rem % DK3_;
    float acc = sc[h][a][0] * vr[0 * C3_ + h * DK3_ + d] +
                sc[h][a][1] * vr[1 * C3_ + h * DK3_ + d] +
                sc[h][a][2] * vr[2 * C3_ + h * DK3_ + d];
    och[(size_t)bl * D_ + c] = (_Float16)(acc * sigm(gr[c]));
  }
}

// ---------------------------------------------------------------- outer-product mean -> mo [B,32,L,64]
__global__ void k_op_outer(const float* __restrict__ tt, float* __restrict__ mo,
                           _Float16* __restrict__ moh) {
  int bl = blockIdx.x; int bb = bl / L_; int l = bl % L_;
  const float* tr = tt + (size_t)bl * 768;
  for (int o = threadIdx.x; o < 32 * 64; o += blockDim.x) {
    int c = o / 64; int rem = o % 64; int i = rem / 8; int j = rem % 8;
    float v = (tr[0 * 256 + c * 8 + i] * tr[0 * 256 + c * 8 + j] +
               tr[1 * 256 + c * 8 + i] * tr[1 * 256 + c * 8 + j] +
               tr[2 * 256 + c * 8 + i] * tr[2 * 256 + c * 8 + j]) * (1.f / 3.f);
    size_t idx = (((size_t)bb * 32 + c) * L_ + l) * 64 + rem;
    mo[idx] = v; moh[idx] = (_Float16)v;
  }
}

// ---------------------------------------------------------------- mt = gelu(mo @ atw^T + atb) (f16 out)
__global__ void k_op_at(const float* __restrict__ mo, const float* atw, const float* atb,
                        _Float16* __restrict__ mth) {
  size_t row = blockIdx.x;              // (b*32+c)*L + m
  __shared__ float mrow[64];
  mrow[threadIdx.x] = mo[row * 64 + threadIdx.x];
  __syncthreads();
  int k2 = threadIdx.x;
  float acc = atb[k2];
  const float* wr = atw + (size_t)k2 * 64;
  for (int k = 0; k < 64; ++k) acc += mrow[k] * wr[k];
  mth[row * 64 + k2] = (_Float16)geluf(acc);
}

// ---------------------------------------------------------------- rs[b,l,m,c] = mo·mt^T   (WMMA, K=64)
__global__ void __launch_bounds__(32)
k_op_rs(const _Float16* __restrict__ moh, const _Float16* __restrict__ mth,
        float* __restrict__ rs) {
  int l0 = blockIdx.x * 16, m0 = blockIdx.y * 16;
  int zc = blockIdx.z;                   // b*32 + c
  int bb = zc / 32, c = zc % 32;
  int lane = threadIdx.x;
  const _Float16* abase = moh + ((size_t)zc * L_ + l0) * 64;
  const _Float16* bbase = mth + ((size_t)zc * L_ + m0) * 64;
  v8f acc = {0.f, 0.f, 0.f, 0.f, 0.f, 0.f, 0.f, 0.f};
#pragma unroll
  for (int kk = 0; kk < 2; ++kk) {
    v16h a = load_a_frag(abase + kk * 32, 64, lane);
    v16h b = load_b_frag(bbase + kk * 32, 64, lane);
    acc = __builtin_amdgcn_wmma_f32_16x16x32_f16(false, a, false, b, (short)0, acc, false, false);
  }
  int n = lane & 15; int rb = (lane >> 4) * 8;
#pragma unroll
  for (int r = 0; r < 8; ++r) {
    int l = l0 + r + rb, m = m0 + n;
    rs[(((size_t)bb * L_ + l) * L_ + m) * 32 + c] = acc[r];
  }
}

// ---------------------------------------------------------------- fused: pairX + rs@t2^T -> LN -> plw -> bias[b,h,i,j]
__global__ void k_bias(const float* __restrict__ pairX, const float* __restrict__ rs,
                       const float* t2w, const float* t2b,
                       const float* lng, const float* lnb,
                       const float* plw, const float* plb, float* __restrict__ biasO) {
  int j = blockIdx.x, i = blockIdx.y, bb = blockIdx.z;
  __shared__ float rsv[32];
  __shared__ float dn[128];
  __shared__ float red[128];
  size_t pbase = ((size_t)bb * L_ + i) * L_ + j;
  if (threadIdx.x < 32) rsv[threadIdx.x] = rs[pbase * 32 + threadIdx.x];
  __syncthreads();
  int t = threadIdx.x;
  float acc = t2b[t];
  const float* wr = t2w + t * 32;
  for (int c = 0; c < 32; ++c) acc += rsv[c] * wr[c];
  float d = pairX[pbase * PAIR_ + t] + acc;
  float s = block_sum(d, red);
  float s2 = block_sum(d * d, red);
  float mean = s / 128.f, inv = rsqrtf(s2 / 128.f - mean * mean + EPS_);
  dn[t] = (d - mean) * inv * lng[t] + lnb[t];
  __syncthreads();
  if (t < H_) {
    float a2 = plb[t];
    const float* pw = plw + t * PAIR_;
    for (int u = 0; u < PAIR_; ++u) a2 += dn[u] * pw[u];
    biasO[(((size_t)bb * H_ + t) * L_ + i) * L_ + j] = a2;
  }
}

// ---------------------------------------------------------------- repack row q/k/v/g into head-major layouts
__global__ void k_repack_qkvg(const float* __restrict__ q4, const float* __restrict__ k4,
                              const float* __restrict__ v4, const float* __restrict__ g4,
                              _Float16* __restrict__ qh, _Float16* __restrict__ kh,
                              float* __restrict__ vv, float* __restrict__ gg) {
  int bl = blockIdx.x; int bb = bl / L_; int l = bl % L_;
  for (int c = threadIdx.x; c < D_; c += blockDim.x) {
    int sp = c / C3_; int rem = c % C3_; int h = rem / DK3_; int d = rem % DK3_;
    size_t hb = (((size_t)(bb * 3 + sp) * H_ + h) * L_ + l);
    qh[hb * 32 + d] = (_Float16)q4[(size_t)bl * D_ + c];
    kh[hb * 32 + d] = (_Float16)k4[(size_t)bl * D_ + c];
    vv[hb * 24 + d] = v4[(size_t)bl * D_ + c];
    gg[hb * 24 + d] = g4[(size_t)bl * D_ + c];
  }
  for (int t = threadIdx.x; t < 3 * H_ * 8; t += blockDim.x) {   // zero K-pad 24..31
    int sp = t / 64; int rem = t % 64; int hh = rem / 8; int dp = 24 + rem % 8;
    size_t hb = (((size_t)(bb * 3 + sp) * H_ + hh) * L_ + l);
    qh[hb * 32 + dp] = (_Float16)0.f;
    kh[hb * 32 + dp] = (_Float16)0.f;
  }
}

// ---------------------------------------------------------------- qk = q·k^T + bias  (WMMA, K padded to 32)
__global__ void __launch_bounds__(32)
k_qk(const _Float16* __restrict__ qh, const _Float16* __restrict__ kh,
     const float* __restrict__ biasO, float* __restrict__ qkO) {
  int i0 = blockIdx.x * 16, j0 = blockIdx.y * 16;
  int z = blockIdx.z;                  // (bb*3+sp)*H + h
  int bb = z / (3 * H_), h = z % H_;
  int lane = threadIdx.x;
  size_t base = (size_t)z * L_;
  v16h a = load_a_frag(qh + (base + i0) * 32, 32, lane);
  v16h b = load_b_frag(kh + (base + j0) * 32, 32, lane);
  v8f acc = {0.f, 0.f, 0.f, 0.f, 0.f, 0.f, 0.f, 0.f};
  acc = __builtin_amdgcn_wmma_f32_16x16x32_f16(false, a, false, b, (short)0, acc, false, false);
  int n = lane & 15; int rb = (lane >> 4) * 8;
  size_t bbase = ((size_t)bb * H_ + h) * L_;
#pragma unroll
  for (int r = 0; r < 8; ++r) {
    int i = i0 + r + rb, j = j0 + n;
    qkO[(base + i) * L_ + j] = acc[r] + biasO[(bbase + i) * L_ + j];
  }
}

// ---------------------------------------------------------------- 5x5 score convolutions, summed over s
__global__ void k_conv(const float* __restrict__ qkO,
                       const float* w0, const float* b0, const float* w1, const float* b1,
                       const float* w2, const float* b2, float* __restrict__ convO) {
  int i0 = blockIdx.x * 16, j0 = blockIdx.y * 16, bb = blockIdx.z;
  __shared__ float inb[3][8][20][20];
  __shared__ float wsh[3][1600];
  for (int t = threadIdx.x; t < 3 * 8 * 400; t += blockDim.x) {
    int sp = t / 3200; int rem = t % 3200; int ci = rem / 400; int p = rem % 400;
    int di = p / 20, dj = p % 20;
    int ii = i0 + di - 2, jj = j0 + dj - 2;
    float v = 0.f;
    if (ii >= 0 && ii < L_ && jj >= 0 && jj < L_)
      v = qkO[((((size_t)bb * 3 + sp) * H_ + ci) * L_ + ii) * L_ + jj];
    inb[sp][ci][di][dj] = v;
  }
  for (int t = threadIdx.x; t < 3 * 1600; t += blockDim.x) {
    int sp = t / 1600; int r = t % 1600;
    wsh[sp][r] = (sp == 0 ? w0[r] : (sp == 1 ? w1[r] : w2[r]));
  }
  __syncthreads();
  for (int t = threadIdx.x; t < 8 * 256; t += blockDim.x) {
    int h = t / 256; int p = t % 256; int di = p / 16, dj = p % 16;
    float acc = b0[h] + b1[h] + b2[h];
    for (int sp = 0; sp < 3; ++sp) {
      const float* ws = &wsh[sp][h * 200];
      for (int ci = 0; ci < 8; ++ci) {
        const float* wc = ws + ci * 25;
        for (int dy = 0; dy < 5; ++dy)
          for (int dx = 0; dx < 5; ++dx)
            acc += inb[sp][ci][di + dy][dj + dx] * wc[dy * 5 + dx];
      }
    }
    convO[(((size_t)bb * H_ + h) * L_ + i0 + di) * L_ + j0 + dj] = acc;
  }
}

// ---------------------------------------------------------------- qk = (qk + conv)/DIV (in place)
__global__ void k_addconv(float* __restrict__ qkO, const float* __restrict__ convO) {
  size_t idx = (size_t)blockIdx.x * blockDim.x + threadIdx.x;
  size_t total = (size_t)B_ * 3 * H_ * L_ * L_;
  if (idx >= total) return;
  size_t jj = idx % L_; size_t t2 = idx / L_; size_t ii = t2 % L_; size_t z = t2 / L_;
  int h = (int)(z % H_); size_t bb = z / (3 * H_);
  float cv = convO[((bb * H_ + h) * L_ + ii) * L_ + jj];
  qkO[idx] = (qkO[idx] + cv) / DIVQ;
}

// ---------------------------------------------------------------- softmax + A·V + gate -> ogh [B,L,576] f16
__global__ void k_softmax_av(const float* __restrict__ qkO, const float* __restrict__ vv,
                             const float* __restrict__ gg, _Float16* __restrict__ ogh) {
  int z = blockIdx.x;                  // ((bb*3+sp)*H+h)*L + i
  int i = z % L_; int zh = z / L_;
  int h = zh % H_; int sp = (zh / H_) % 3; int bb = zh / (3 * H_);
  __shared__ float p[L_];
  __shared__ float red[128];
  const float* row = qkO + (size_t)z * L_;
  float m = -1e30f;
  for (int j = threadIdx.x; j < L_; j += blockDim.x) m = fmaxf(m, row[j]);
  m = block_max(m, red);
  float s = 0.f;
  for (int j = threadIdx.x; j < L_; j += blockDim.x) { float e = expf(row[j] - m); p[j] = e; s += e; }
  s = block_sum(s, red);
  float invs = 1.f / s;
  __syncthreads();
  if (threadIdx.x < DK3_) {
    int d = threadIdx.x;
    const float* vb2 = vv + (size_t)zh * L_ * 24;
    float acc = 0.f;
    for (int j = 0; j < L_; ++j) acc += p[j] * vb2[j * 24 + d];
    acc *= invs;
    acc *= sigm(gg[((size_t)zh * L_ + i) * 24 + d]);
    ogh[((size_t)bb * L_ + i) * D_ + (sp * H_ + h) * DK3_ + d] = (_Float16)acc;
  }
}

// ---------------------------------------------------------------- fused pair head, one block per (i,b) row
// rsm=gelu(mean_s qk) -> prw -> LN -> pf FFN (WMMA x2 from LDS-resident f16 weights)
__global__ void __launch_bounds__(512)
k_pair_head(const float* __restrict__ qkO,
            const float* __restrict__ prw, const float* __restrict__ prb,
            const float* __restrict__ lng, const float* __restrict__ lnb,
            const _Float16* __restrict__ w1h, const float* __restrict__ b1f,
            const _Float16* __restrict__ w2h, const float* __restrict__ b2f,
            float* __restrict__ pairO) {
  int i = blockIdx.x; int bb = blockIdx.y;
  int wave = threadIdx.x >> 5, lane = threadIdx.x & 31;
  __shared__ _Float16 w1s[256 * PAIR_];            // 64 KB  (pf_w1, f16)
  __shared__ _Float16 w2s[PAIR_ * 256];            // 64 KB  (pf_w2, f16)
  __shared__ _Float16 As[16][PAIR_];
  __shared__ _Float16 Hs[16][256];
  __shared__ float prwS[PAIR_ * H_];
  __shared__ float prbS[PAIR_], lngS[PAIR_], lnbS[PAIR_], b2S[PAIR_], b1S[256];

  // ---- stage weights into LDS: TDM bulk copy + small cooperative copies ----
#if HAVE_TDM
  if (wave == 0 && lane == 0) {
    tdm_stage(w1s, w1h, 256 * PAIR_);
    tdm_stage(w2s, w2h, PAIR_ * 256);
  }
#endif
  for (int t = threadIdx.x; t < PAIR_ * H_; t += blockDim.x) prwS[t] = prw[t];
  for (int t = threadIdx.x; t < 256; t += blockDim.x) b1S[t] = b1f[t];
  if (threadIdx.x < PAIR_) {
    prbS[threadIdx.x] = prb[threadIdx.x];
    lngS[threadIdx.x] = lng[threadIdx.x];
    lnbS[threadIdx.x] = lnb[threadIdx.x];
    b2S[threadIdx.x]  = b2f[threadIdx.x];
  }
#if HAVE_TDM
  if (wave == 0) __builtin_amdgcn_s_wait_tensorcnt(0);
#else
  for (int t = threadIdx.x; t < 256 * PAIR_; t += blockDim.x) w1s[t] = w1h[t];
  for (int t = threadIdx.x; t < PAIR_ * 256; t += blockDim.x) w2s[t] = w2h[t];
#endif
  __syncthreads();

  for (int j0 = 0; j0 < L_; j0 += 16) {
    // ---- phase 1: wave w builds LN'd pair row for j = j0+w (f16 into As) ----
    {
      int j = j0 + wave;
      float rsmv = 0.f;
      if (lane < H_) {
        float a = 0.f;
        for (int sp = 0; sp < 3; ++sp)
          a += qkO[((((size_t)bb * 3 + sp) * H_ + lane) * L_ + i) * L_ + j];
        rsmv = geluf(a * (1.f / 3.f));
      }
      float dv[4];
#pragma unroll
      for (int r = 0; r < 4; ++r) {
        int t = lane * 4 + r;
        float acc = prbS[t];
#pragma unroll
        for (int h = 0; h < H_; ++h) acc += __shfl(rsmv, h, 32) * prwS[t * H_ + h];
        dv[r] = acc;
      }
      float s = dv[0] + dv[1] + dv[2] + dv[3];
      float s2 = dv[0] * dv[0] + dv[1] * dv[1] + dv[2] * dv[2] + dv[3] * dv[3];
      for (int off = 16; off > 0; off >>= 1) {
        s += __shfl_xor(s, off, 32);
        s2 += __shfl_xor(s2, off, 32);
      }
      float mean = s / 128.f, inv = rsqrtf(s2 / 128.f - mean * mean + EPS_);
#pragma unroll
      for (int r = 0; r < 4; ++r) {
        int t = lane * 4 + r;
        As[wave][t] = (_Float16)((dv[r] - mean) * inv * lngS[t] + lnbS[t]);
      }
    }
    __syncthreads();
    // ---- GEMM1: h = gelu(A[16x128] @ w1^T + b1), each wave one 16-wide N tile ----
    {
      int n0 = wave * 16;
      v8f acc = {0.f, 0.f, 0.f, 0.f, 0.f, 0.f, 0.f, 0.f};
#pragma unroll
      for (int kk = 0; kk < 4; ++kk) {
        v16h a = load_a_frag(&As[0][kk * 32], PAIR_, lane);
        v16h b = load_b_frag(&w1s[n0 * PAIR_ + kk * 32], PAIR_, lane);
        acc = __builtin_amdgcn_wmma_f32_16x16x32_f16(false, a, false, b, (short)0, acc, false, false);
      }
      int n = n0 + (lane & 15); int rb = (lane >> 4) * 8;
#pragma unroll
      for (int r = 0; r < 8; ++r) {
        int m = r + rb;
        Hs[m][n] = (_Float16)geluf(acc[r] + b1S[n]);
      }
    }
    __syncthreads();
    // ---- GEMM2: out = H[16x256] @ w2^T + b2, waves 0..7 ----
    if (wave < 8) {
      int u0 = wave * 16;
      v8f acc = {0.f, 0.f, 0.f, 0.f, 0.f, 0.f, 0.f, 0.f};
#pragma unroll
      for (int kk = 0; kk < 8; ++kk) {
        v16h a = load_a_frag(&Hs[0][kk * 32], 256, lane);
        v16h b = load_b_frag(&w2s[u0 * 256 + kk * 32], 256, lane);
        acc = __builtin_amdgcn_wmma_f32_16x16x32_f16(false, a, false, b, (short)0, acc, false, false);
      }
      int u = u0 + (lane & 15); int rb = (lane >> 4) * 8;
#pragma unroll
      for (int r = 0; r < 8; ++r) {
        int m = r + rb;
        pairO[(((size_t)bb * L_ + i) * L_ + (j0 + m)) * PAIR_ + u] = acc[r] + b2S[u];
      }
    }
  }
}

// ================================================================ host launcher
extern "C" void kernel_launch(void* const* d_in, const int* in_sizes, int n_in,
                              void* d_out, int out_size, void* d_ws, size_t ws_size,
                              hipStream_t stream) {
  (void)in_sizes; (void)n_in; (void)out_size; (void)ws_size;
  const float* x      = (const float*)d_in[0];
  const float* pairX  = (const float*)d_in[1];
  const float* P[65];
  for (int k = 0; k < 65; ++k) P[k] = (const float*)d_in[2 + k];
  //  0 cf_ln_g  1 cf_ln_b  2 cf_exp_w 3 cf_exp_b 4 cf_dw_w 5 cf_pw_w 6 cf_pw_b
  //  7 col_g 8 col_ln_g 9 col_ln_b 10 col_qw 11 col_qb 12 col_kw 13 col_kb
  // 14 col_vw 15 col_vb 16 col_gw 17 col_gb 18 col_ow 19 col_ob
  // 20 op_ln_g 21 op_ln_b 22 op_t1w 23 op_t1b 24 op_atw 25 op_atb 26 op_t2w 27 op_t2b
  // 28 row_g 29 row_ln1_g 30 row_ln1_b 31 row_ln2_g 32 row_ln2_b
  // 33 row_qw 34 row_qb 35 row_kw 36 row_kb 37 row_vw 38 row_vb 39 row_gw 40 row_gb
  // 41 row_c0w 42 row_c0b 43 row_c1w 44 row_c1b 45 row_c2w 46 row_c2b
  // 47 row_plw 48 row_plb 49 row_prw 50 row_prb 51 row_ow 52 row_ob
  // 53 ff_ln_g 54 ff_ln_b 55 ff_w1 56 ff_b1 57 ff_w2 58 ff_b2
  // 59 pf_ln_g 60 pf_ln_b 61 pf_w1 62 pf_b1 63 pf_w2 64 pf_b2

  const int    BL   = B_ * L_;                       // 768 tokens (M, multiple of 16)
  const size_t N_X  = (size_t)BL * D_;
  const size_t N_H1 = (size_t)BL * EXP_;
  const size_t N_MO = (size_t)B_ * 32 * L_ * 64;
  const size_t N_RS = (size_t)B_ * L_ * L_ * 32;
  const size_t N_BI = (size_t)B_ * H_ * L_ * L_;
  const size_t N_QH = (size_t)B_ * 3 * H_ * L_ * 32;
  const size_t N_VV = (size_t)B_ * 3 * H_ * L_ * 24;
  const size_t N_QK = (size_t)B_ * 3 * H_ * L_ * L_;

  char* wp = (char*)d_ws;
  auto alloc = [&](size_t bytes) -> void* {
    void* p = (void*)wp;
    wp += (bytes + 255) & ~(size_t)255;
    return p;
  };
  // f32 buffers
  float*    xln  = (float*)alloc(N_X * 4);
  float*    x1   = (float*)alloc(N_X * 4);
  float*    x2   = (float*)alloc(N_X * 4);
  float*    x3   = (float*)alloc(N_X * 4);
  float*    q4   = (float*)alloc(N_X * 4);
  float*    k4   = (float*)alloc(N_X * 4);
  float*    v4   = (float*)alloc(N_X * 4);
  float*    g4   = (float*)alloc(N_X * 4);
  float*    tt   = (float*)alloc((size_t)BL * 768 * 4);
  float*    mo   = (float*)alloc(N_MO * 4);
  float*    rs   = (float*)alloc(N_RS * 4);
  float*    bias = (float*)alloc(N_BI * 4);
  float*    vv   = (float*)alloc(N_VV * 4);
  float*    gg   = (float*)alloc(N_VV * 4);
  float*    qk   = (float*)alloc(N_QK * 4);
  float*    cv   = (float*)alloc(N_BI * 4);
  // f16 buffers
  _Float16* Acf    = (_Float16*)alloc((size_t)BL * KCF_ * 2);
  _Float16* h1h    = (_Float16*)alloc(N_H1 * 2);
  _Float16* h2h    = (_Float16*)alloc(N_H1 * 2);
  _Float16* ylnh   = (_Float16*)alloc(N_X * 2);
  _Float16* och    = (_Float16*)alloc(N_X * 2);
  _Float16* oplnh  = (_Float16*)alloc(N_X * 2);
  _Float16* rowlnh = (_Float16*)alloc(N_X * 2);
  _Float16* ogh    = (_Float16*)alloc(N_X * 2);
  _Float16* fflnh  = (_Float16*)alloc(N_X * 2);
  _Float16* ffh    = (_Float16*)alloc(N_H1 * 2);
  _Float16* moh    = (_Float16*)alloc(N_MO * 2);
  _Float16* mth    = (_Float16*)alloc(N_MO * 2);
  _Float16* qhB    = (_Float16*)alloc(N_QH * 2);
  _Float16* khB    = (_Float16*)alloc(N_QH * 2);
  _Float16* w1h16  = (_Float16*)alloc((size_t)256 * PAIR_ * 2);
  _Float16* w2h16  = (_Float16*)alloc((size_t)PAIR_ * 256 * 2);

  float* x_out    = (float*)d_out;
  float* pair_out = (float*)d_out + N_X;

  const int MT = BL / 16;                    // 48 M-tiles
  // ---- pf weights -> f16 (once per call) ----
  k_f32h<<<(256 * PAIR_ + 255) / 256, 256, 0, stream>>>(P[61], w1h16, 256 * PAIR_);
  k_f32h<<<(PAIR_ * 256 + 255) / 256, 256, 0, stream>>>(P[63], w2h16, PAIR_ * 256);
  // ---- x path: ConvFFN (im2col GEMM + depthwise + pointwise GEMM) ----
  k_ln<<<BL, 256, 0, stream>>>(x, P[0], P[1], xln, nullptr, D_);
  k_stage_cf<<<BL, 256, 0, stream>>>(xln, Acf);
  k_gemm<<<dim3(MT, EXP_ / 16), 32, 0, stream>>>(Acf, KCF_, P[2], KCF_, P[3],
                                                 nullptr, 0, nullptr,
                                                 nullptr, h1h, EXP_, KCF_, 1);
  k_dw<<<BL, 256, 0, stream>>>(h1h, P[4], h2h);
  k_gemm<<<dim3(MT, D_ / 16), 32, 0, stream>>>(h2h, EXP_, P[5], EXP_, P[6],
                                               x, D_, nullptr,
                                               x1, nullptr, D_, EXP_, 0);
  // ---- column attention ----
  k_ln<<<BL, 256, 0, stream>>>(x1, P[8], P[9], nullptr, ylnh, D_);
  for (int sp = 0; sp < 3; ++sp) {
    const _Float16* Abase = ylnh + sp * C3_;
    float* outs[4] = {q4, k4, v4, g4};
    const float* Ws[4] = {P[10], P[12], P[14], P[16]};
    const float* Bs[4] = {P[11], P[13], P[15], P[17]};
    for (int mat = 0; mat < 4; ++mat)
      k_gemm<<<dim3(MT, C3_ / 16), 32, 0, stream>>>(Abase, D_, Ws[mat], C3_, Bs[mat],
                                                    nullptr, 0, nullptr,
                                                    outs[mat] + sp * C3_, nullptr, D_, C3_, 0);
  }
  k_col_small<<<BL, 128, 0, stream>>>(q4, k4, v4, g4, och);
  k_gemm<<<dim3(MT, D_ / 16), 32, 0, stream>>>(och, D_, P[18], D_, P[19],
                                               x1, D_, P[7],
                                               x2, nullptr, D_, D_, 0);
  // ---- outer-product pair update (feeds bias) ----
  k_ln<<<BL, 256, 0, stream>>>(x2, P[20], P[21], nullptr, oplnh, D_);
  for (int sp = 0; sp < 3; ++sp)
    k_gemm<<<dim3(MT, 256 / 16), 32, 0, stream>>>(oplnh + sp * C3_, D_, P[22], C3_, P[23],
                                                  nullptr, 0, nullptr,
                                                  tt + sp * 256, nullptr, 768, C3_, 0);
  k_op_outer<<<BL, 256, 0, stream>>>(tt, mo, moh);
  k_op_at<<<B_ * 32 * L_, 64, 0, stream>>>(mo, P[24], P[25], mth);
  k_op_rs<<<dim3(L_ / 16, L_ / 16, B_ * 32), 32, 0, stream>>>(moh, mth, rs);
  k_bias<<<dim3(L_, L_, B_), 128, 0, stream>>>(pairX, rs, P[26], P[27], P[31], P[32],
                                               P[47], P[48], bias);
  // ---- row attention ----
  k_ln<<<BL, 256, 0, stream>>>(x2, P[29], P[30], nullptr, rowlnh, D_);
  for (int sp = 0; sp < 3; ++sp) {
    const _Float16* Abase = rowlnh + sp * C3_;
    float* outs[4] = {q4, k4, v4, g4};
    const float* Ws[4] = {P[33], P[35], P[37], P[39]};
    const float* Bs[4] = {P[34], P[36], P[38], P[40]};
    for (int mat = 0; mat < 4; ++mat)
      k_gemm<<<dim3(MT, C3_ / 16), 32, 0, stream>>>(Abase, D_, Ws[mat], C3_, Bs[mat],
                                                    nullptr, 0, nullptr,
                                                    outs[mat] + sp * C3_, nullptr, D_, C3_, 0);
  }
  k_repack_qkvg<<<BL, 256, 0, stream>>>(q4, k4, v4, g4, qhB, khB, vv, gg);
  k_qk<<<dim3(L_ / 16, L_ / 16, B_ * 3 * H_), 32, 0, stream>>>(qhB, khB, bias, qk);
  k_conv<<<dim3(L_ / 16, L_ / 16, B_), 256, 0, stream>>>(qk, P[41], P[42], P[43], P[44],
                                                         P[45], P[46], cv);
  {
    int blocks = (int)((N_QK + 255) / 256);
    k_addconv<<<blocks, 256, 0, stream>>>(qk, cv);
  }
  k_softmax_av<<<B_ * 3 * H_ * L_, 128, 0, stream>>>(qk, vv, gg, ogh);
  k_gemm<<<dim3(MT, D_ / 16), 32, 0, stream>>>(ogh, D_, P[51], D_, P[52],
                                               x2, D_, P[28],
                                               x3, nullptr, D_, D_, 0);
  // ---- x feed-forward (writes x part of d_out) ----
  k_ln<<<BL, 256, 0, stream>>>(x3, P[53], P[54], nullptr, fflnh, D_);
  k_gemm<<<dim3(MT, EXP_ / 16), 32, 0, stream>>>(fflnh, D_, P[55], D_, P[56],
                                                 nullptr, 0, nullptr,
                                                 nullptr, ffh, EXP_, D_, 1);
  k_gemm<<<dim3(MT, D_ / 16), 32, 0, stream>>>(ffh, EXP_, P[57], EXP_, P[58],
                                               x3, D_, nullptr,
                                               x_out, nullptr, D_, EXP_, 0);
  // ---- fused pair head (writes pair part of d_out) ----
  k_pair_head<<<dim3(L_, B_), 512, 0, stream>>>(qk, P[49], P[50], P[59], P[60],
                                                w1h16, P[62], w2h16, P[64], pair_out);
}
